// MyLstmModel_80384607912440
// MI455X (gfx1250) — matrix operand order repaired and checked by reading, not compiled, not run on hardware
//
#include <hip/hip_runtime.h>

typedef _Float16 half_t;
typedef __attribute__((ext_vector_type(16))) _Float16 v16h;
typedef __attribute__((ext_vector_type(8)))  _Float16 v8h;
typedef __attribute__((ext_vector_type(8)))  float    v8f;

#define B_SZ 1024
#define T_SZ 128

__device__ __forceinline__ float sigmoid_f(float x) {
  return 1.0f / (1.0f + __expf(-x));
}
__device__ __forceinline__ float tanh_f(float x) {
  return 2.0f / (1.0f + __expf(-2.0f * x)) - 1.0f;
}

// ---------------------------------------------------------------------------
// f32 -> f16 with column zero-padding: out has FP cols, in has F cols.
// ---------------------------------------------------------------------------
__global__ void k_convert_pad(const float* __restrict__ in, half_t* __restrict__ out,
                              int rows, int F, int FP) {
  long long i = (long long)blockIdx.x * blockDim.x + threadIdx.x;
  long long tot = (long long)rows * FP;
  if (i >= tot) return;
  int c = (int)(i % FP);
  long long r = i / FP;
  out[i] = (c < F) ? (half_t)in[r * (long long)F + c] : (half_t)0.0f;
}

// ---------------------------------------------------------------------------
// Pack row-major W[K][N] (f32) into WMMA B-fragments (f16), zero-padding K up
// to a multiple of 32.  Layout per (chunk,tile): 32 lanes x 16 halves, where
// lane 0-15 holds col = tile*16+lane, K = chunk*32 + {0..7, 16..23} (elements 0..15),
// lane 16-31 holds col = tile*16+lane-16, K = chunk*32 + {8..15, 24..31}.
// One 32-thread block per (chunk, tile);  grid = Kchunks * (N/16).
// ---------------------------------------------------------------------------
__global__ void k_pack_b(const float* __restrict__ W, half_t* __restrict__ out,
                         int K, int N) {
  int Nt    = N >> 4;
  int lin   = blockIdx.x;           // chunk*Nt + tile
  int tile  = lin % Nt;
  int chunk = lin / Nt;
  int lane  = threadIdx.x;
  int col   = (tile << 4) + (lane & 15);
  int kOff  = (chunk << 5) + ((lane >> 4) << 3);
  half_t* o = out + ((size_t)lin * 32 + lane) * 16;
#pragma unroll
  for (int j = 0; j < 16; ++j) {
    int k = kOff + j + ((j >> 3) << 3);
    o[j] = (k < K) ? (half_t)W[(size_t)k * N + col] : (half_t)0.0f;
  }
}

// ---------------------------------------------------------------------------
// Fused LSTM recurrence (one direction).
//   U  : hidden units of this direction (16 | 32 | 64)
//   FP : padded input feature width (multiple of 32); xin is [B][T][FP] f16
//   wk : packed B-fragments of Wk (FP x 4U)  -> staged into LDS per block
//   wr : packed B-fragments of Wr (KHP x 4U) -> preloaded into VGPRs
//   hout: [B][T][hstride] f16, this direction writes cols [colOff, colOff+U)
//   dir: 0 = forward (step s uses time s), 1 = backward (step s uses T-1-s,
//        which also re-reverses the output as Keras Bidirectional does).
// One block = 16 batch rows; NW = U/16 waves; wave w owns units [16w,16w+16)
// and computes gates i,f,g,o for them (N-tiles g*NW+w).  c-state in VGPRs.
// ---------------------------------------------------------------------------
template <int U, int FP>
__global__ __launch_bounds__(U * 2, 1)
void k_lstm_rec(const half_t* __restrict__ xin,
                const half_t* __restrict__ wk,
                const half_t* __restrict__ wr,
                const float*  __restrict__ bias,   // [4U], order i,f,g,o
                half_t* __restrict__ hout,
                int hstride, int colOff, int dir) {
  constexpr int NW  = U / 16;        // waves per block
  constexpr int KX  = FP / 32;       // input K chunks
  constexpr int KH  = (U + 31) / 32; // hidden K chunks
  constexpr int KHP = KH * 32;       // padded hidden width in LDS
  constexpr int NT  = (4 * U) / 16;  // gate N-tiles total
  constexpr int WKH = KX * NT * 512; // halves in packed Wk

  __shared__ alignas(16) half_t hbuf[16][KHP];
  __shared__ alignas(32) half_t wkbuf[WKH];

  const int tid   = threadIdx.x;
  const int w     = tid >> 5;
  const int lane  = tid & 31;
  const int row   = lane & 15;
  const int hi    = lane >> 4;
  const int bBase = blockIdx.x << 4;

  // Stage packed Wk fragments into LDS (shared by all waves).
  for (int i = tid * 8; i < WKH; i += NW * 32 * 8)
    *(v8h*)&wkbuf[i] = *(const v8h*)&wk[i];
  // h0 = 0 (including the K-padding columns, which stay zero forever)
  for (int i = tid; i < 16 * KHP; i += NW * 32) (&hbuf[0][0])[i] = (half_t)0.0f;
  __syncthreads();

  // Recurrent weight fragments live in registers (hot, latency-critical path).
  v16h wrf[KH][4];
#pragma unroll
  for (int c = 0; c < KH; ++c)
#pragma unroll
    for (int g = 0; g < 4; ++g) {
      int tile = g * NW + w;
      wrf[c][g] = *(const v16h*)(wr + ((size_t)(c * NT + tile) * 32 + lane) * 16);
    }
  float bv[4];
#pragma unroll
  for (int g = 0; g < 4; ++g) bv[g] = bias[g * U + (w << 4) + row];

  float cst[8];
#pragma unroll
  for (int j = 0; j < 8; ++j) cst[j] = 0.0f;

  for (int s = 0; s < T_SZ; ++s) {
    const int t = dir ? (T_SZ - 1 - s) : s;

    v8f acc[4];
#pragma unroll
    for (int g = 0; g < 4; ++g)
#pragma unroll
      for (int j = 0; j < 8; ++j) acc[g][j] = bv[g];

    // z += x_t @ Wk   (A from global f16 x, B fragments from LDS)
    const half_t* xrow = xin + ((size_t)(bBase + row) * T_SZ + t) * FP + (hi << 3);
#pragma unroll
    for (int c = 0; c < KX; ++c) {
      union { v16h v; v8h h[2]; } a;
      a.h[0] = *(const v8h*)(xrow + c * 32);
      a.h[1] = *(const v8h*)(xrow + c * 32 + 16);
      v16h bfr[4];
#pragma unroll
      for (int g = 0; g < 4; ++g)
        bfr[g] = *(const v16h*)&wkbuf[((c * NT + g * NW + w) * 32 + lane) * 16];
#pragma unroll
      for (int g = 0; g < 4; ++g)
        acc[g] = __builtin_amdgcn_wmma_f32_16x16x32_f16(
            false, a.v, false, bfr[g], (short)0, acc[g], false, false);
    }

    // z += h_{t-1} @ Wr   (A from LDS, B fragments already in VGPRs)
    const half_t* hrow = &hbuf[row][hi << 3];
#pragma unroll
    for (int c = 0; c < KH; ++c) {
      union { v16h v; v8h h[2]; } a;
      a.h[0] = *(const v8h*)(hrow + c * 32);
      a.h[1] = *(const v8h*)(hrow + c * 32 + 16);
#pragma unroll
      for (int g = 0; g < 4; ++g)
        acc[g] = __builtin_amdgcn_wmma_f32_16x16x32_f16(
            false, a.v, false, wrf[c][g], (short)0, acc[g], false, false);
    }

    // Gate math: lane owns unit u for batch rows hi*8 .. hi*8+7; c in VGPRs.
    float hnew[8];
#pragma unroll
    for (int j = 0; j < 8; ++j) {
      float ig = sigmoid_f(acc[0][j]);
      float fg = sigmoid_f(acc[1][j]);
      float gg = tanh_f(acc[2][j]);
      float og = sigmoid_f(acc[3][j]);
      cst[j]  = fg * cst[j] + ig * gg;
      hnew[j] = og * tanh_f(cst[j]);
    }

    __syncthreads();  // all waves done reading hbuf this step
    const int u = (w << 4) + row;
#pragma unroll
    for (int j = 0; j < 8; ++j) {
      int m = (hi << 3) + j;
      half_t hh = (half_t)hnew[j];
      hbuf[m][u] = hh;
      hout[((size_t)(bBase + m) * T_SZ + t) * hstride + colOff + u] = hh;
    }
    __syncthreads();  // writes visible before next step's reads
  }
}

// ---------------------------------------------------------------------------
// Head: concat(final fwd h3 @ t=T-1 cols 0..15, final bwd h3 @ t=0 cols 16..31)
// -> relu dense(32->8) -> sigmoid dense(8->3)
// ---------------------------------------------------------------------------
__global__ void k_head(const half_t* __restrict__ h3,
                       const float* __restrict__ d3w, const float* __restrict__ d3b,
                       const float* __restrict__ cw,  const float* __restrict__ cb,
                       float* __restrict__ out) {
  int b = blockIdx.x * blockDim.x + threadIdx.x;
  if (b >= B_SZ) return;
  float h[32];
#pragma unroll
  for (int u = 0; u < 16; ++u)
    h[u] = (float)h3[((size_t)b * T_SZ + (T_SZ - 1)) * 32 + u];
#pragma unroll
  for (int u = 0; u < 16; ++u)
    h[16 + u] = (float)h3[((size_t)b * T_SZ + 0) * 32 + 16 + u];
  float d[8];
#pragma unroll
  for (int j = 0; j < 8; ++j) {
    float s = d3b[j];
#pragma unroll
    for (int u = 0; u < 32; ++u) s += h[u] * d3w[u * 8 + j];
    d[j] = s > 0.0f ? s : 0.0f;
  }
#pragma unroll
  for (int k = 0; k < 3; ++k) {
    float s = cb[k];
#pragma unroll
    for (int j = 0; j < 8; ++j) s += d[j] * cw[j * 3 + k];
    out[b * 3 + k] = sigmoid_f(s);
  }
}

// ---------------------------------------------------------------------------
extern "C" void kernel_launch(void* const* d_in, const int* in_sizes, int n_in,
                              void* d_out, int out_size, void* d_ws, size_t ws_size,
                              hipStream_t stream) {
  (void)in_sizes; (void)n_in; (void)out_size; (void)ws_size;

  const float* x     = (const float*)d_in[0];
  const float* w1f_k = (const float*)d_in[1];
  const float* w1f_r = (const float*)d_in[2];
  const float* w1f_b = (const float*)d_in[3];
  const float* w1b_k = (const float*)d_in[4];
  const float* w1b_r = (const float*)d_in[5];
  const float* w1b_b = (const float*)d_in[6];
  const float* w2f_k = (const float*)d_in[7];
  const float* w2f_r = (const float*)d_in[8];
  const float* w2f_b = (const float*)d_in[9];
  const float* w2b_k = (const float*)d_in[10];
  const float* w2b_r = (const float*)d_in[11];
  const float* w2b_b = (const float*)d_in[12];
  const float* w3f_k = (const float*)d_in[13];
  const float* w3f_r = (const float*)d_in[14];
  const float* w3f_b = (const float*)d_in[15];
  const float* w3b_k = (const float*)d_in[16];
  const float* w3b_r = (const float*)d_in[17];
  const float* w3b_b = (const float*)d_in[18];
  const float* d3_w  = (const float*)d_in[19];
  const float* d3_b  = (const float*)d_in[20];
  const float* cls_w = (const float*)d_in[21];
  const float* cls_b = (const float*)d_in[22];

  char* ws = (char*)d_ws;
  size_t off = 0;
  auto alloc = [&](size_t bytes) -> void* {
    off = (off + 255) & ~(size_t)255;
    void* p = ws + off;
    off += bytes;
    return p;
  };
  const size_t BT = (size_t)B_SZ * T_SZ;
  half_t* xh = (half_t*)alloc(BT * 96  * sizeof(half_t));  // x padded 78->96
  half_t* h1 = (half_t*)alloc(BT * 128 * sizeof(half_t));
  half_t* h2 = (half_t*)alloc(BT * 64  * sizeof(half_t));
  half_t* h3 = (half_t*)alloc(BT * 32  * sizeof(half_t));

  auto packsz = [](int K, int N) -> size_t {
    return (size_t)((K + 31) / 32) * (N / 16) * 32 * 16 * sizeof(half_t);
  };
  half_t* p1f_k = (half_t*)alloc(packsz(78, 256));
  half_t* p1f_r = (half_t*)alloc(packsz(64, 256));
  half_t* p1b_k = (half_t*)alloc(packsz(78, 256));
  half_t* p1b_r = (half_t*)alloc(packsz(64, 256));
  half_t* p2f_k = (half_t*)alloc(packsz(128, 128));
  half_t* p2f_r = (half_t*)alloc(packsz(32, 128));
  half_t* p2b_k = (half_t*)alloc(packsz(128, 128));
  half_t* p2b_r = (half_t*)alloc(packsz(32, 128));
  half_t* p3f_k = (half_t*)alloc(packsz(64, 64));
  half_t* p3f_r = (half_t*)alloc(packsz(16, 64));
  half_t* p3b_k = (half_t*)alloc(packsz(64, 64));
  half_t* p3b_r = (half_t*)alloc(packsz(16, 64));

  auto pack = [&](const float* W, half_t* dst, int K, int N) {
    int grid = ((K + 31) / 32) * (N / 16);
    k_pack_b<<<dim3(grid), dim3(32), 0, stream>>>(W, dst, K, N);
  };
  pack(w1f_k, p1f_k, 78, 256);  pack(w1f_r, p1f_r, 64, 256);
  pack(w1b_k, p1b_k, 78, 256);  pack(w1b_r, p1b_r, 64, 256);
  pack(w2f_k, p2f_k, 128, 128); pack(w2f_r, p2f_r, 32, 128);
  pack(w2b_k, p2b_k, 128, 128); pack(w2b_r, p2b_r, 32, 128);
  pack(w3f_k, p3f_k, 64, 64);   pack(w3f_r, p3f_r, 16, 64);
  pack(w3b_k, p3b_k, 64, 64);   pack(w3b_r, p3b_r, 16, 64);

  {
    long long tot = (long long)BT * 96;
    int blocks = (int)((tot + 255) / 256);
    k_convert_pad<<<dim3(blocks), dim3(256), 0, stream>>>(x, xh, (int)BT, 78, 96);
  }

  const int NB = B_SZ / 16;  // 64 blocks, 16 batch rows each
  // Layer 1: U=64 per dir, input FP=96, concat out stride 128
  k_lstm_rec<64, 96><<<dim3(NB), dim3(128), 0, stream>>>(xh, p1f_k, p1f_r, w1f_b, h1, 128, 0, 0);
  k_lstm_rec<64, 96><<<dim3(NB), dim3(128), 0, stream>>>(xh, p1b_k, p1b_r, w1b_b, h1, 128, 64, 1);
  // Layer 2: U=32, input FP=128, out stride 64
  k_lstm_rec<32, 128><<<dim3(NB), dim3(64), 0, stream>>>(h1, p2f_k, p2f_r, w2f_b, h2, 64, 0, 0);
  k_lstm_rec<32, 128><<<dim3(NB), dim3(64), 0, stream>>>(h1, p2b_k, p2b_r, w2b_b, h2, 64, 32, 1);
  // Layer 3: U=16, input FP=64, out stride 32 (full sequence; head picks finals)
  k_lstm_rec<16, 64><<<dim3(NB), dim3(32), 0, stream>>>(h2, p3f_k, p3f_r, w3f_b, h3, 32, 0, 0);
  k_lstm_rec<16, 64><<<dim3(NB), dim3(32), 0, stream>>>(h2, p3b_k, p3b_r, w3b_b, h3, 32, 16, 1);

  k_head<<<dim3(B_SZ / 256), dim3(256), 0, stream>>>(h3, d3_w, d3_b, cls_w, cls_b, (float*)d_out);
}
